// ExtendedMultiHeadAttention_87943750352938
// MI455X (gfx1250) — matrix operand order repaired and checked by reading, not compiled, not run on hardware
//
#include <hip/hip_runtime.h>
#include <cstdint>
#include <cstddef>

// Problem constants
#define DIM    2048
#define NH     32
#define NKV    8
#define HD     64
#define NB     4
#define TT     1024
#define BT     4096     // NB * TT
#define KV_DIM 512      // NKV * HD
// softmax scale folded into Q projection: HD^-0.5 * log2(e)
#define QSCALE 0.18033688011112042f

typedef __attribute__((ext_vector_type(16))) __bf16 v16bf;
typedef __attribute__((ext_vector_type(8)))  __bf16 v8bf;
typedef __attribute__((ext_vector_type(8)))  float  v8f;

union ABf { v16bf v; v8bf h[2]; };
union F8  { v8f v; float f[8]; };
union B8  { v8bf v; __bf16 e[8]; };

__device__ inline __bf16 f2bf(float x) {
  union { float f; uint32_t u; } in; in.f = x;
  uint32_t r = in.u + 0x7FFFu + ((in.u >> 16) & 1u);
  union { uint16_t u; __bf16 b; } out; out.u = (uint16_t)(r >> 16);
  return out.b;
}

__device__ inline v8f wmma_bf16(v16bf a, v16bf b, v8f c) {
  return __builtin_amdgcn_wmma_f32_16x16x32_bf16(
      /*neg_a=*/false, a, /*neg_b=*/false, b,
      /*c_mod=*/(short)0, c, /*reuse_a=*/false, /*reuse_b=*/false);
}

// ------------------------------------------------------------ converts
__global__ void cvt_bf16_kernel(const float* __restrict__ in,
                                __bf16* __restrict__ out, int n) {
  int i = blockIdx.x * blockDim.x + threadIdx.x;
  int stride = gridDim.x * blockDim.x;
  for (; i < n; i += stride) out[i] = f2bf(in[i]);
}

// fp32 [R][C] -> bf16 [C][R]  (one-time weight transpose)
__global__ __launch_bounds__(256)
void cvt_t_bf16_kernel(const float* __restrict__ in, __bf16* __restrict__ out,
                       int R, int C) {
  __shared__ float tile[32][33];
  const int r0 = blockIdx.y * 32, c0 = blockIdx.x * 32;
  const int tx = threadIdx.x & 31, ty = threadIdx.x >> 5;
#pragma unroll
  for (int i = ty; i < 32; i += 8)
    tile[i][tx] = in[(size_t)(r0 + i) * C + c0 + tx];
  __syncthreads();
#pragma unroll
  for (int i = ty; i < 32; i += 8)
    out[(size_t)(c0 + i) * R + r0 + tx] = f2bf(tile[tx][i]);
}

// ------------------------------------------------------------ GEMM
// C[M,N] = (A[M,K] @ Wt[N,K]^T + bias) * oscale.  No LDS: fragments load
// directly from global (L2-resident on MI455X).  Block 256 thr = 8 waves;
// block tile 128x128; wave tile 32x64 (2x4 WMMA per K-step).
// mode: 0 = fp32 row-major, 1 = bf16 row-major, 2 = bf16 V-transposed
//       (out[b][kvh][dd][s], packed 8-wide stores along s)
__global__ __launch_bounds__(256)
void gemm_wmma_kernel(const __bf16* __restrict__ A, const __bf16* __restrict__ Wt,
                      const float* __restrict__ bias, float* __restrict__ Cf,
                      __bf16* __restrict__ Cb, int M, int N, int K, int mode,
                      float oscale)
{
  const int tid  = threadIdx.x;
  const int lane = tid & 31;
  const int l16  = lane & 15;
  const int lhi  = lane >> 4;
  const int wave = tid >> 5;
  const int wm   = (wave & 3) * 32;   // 2 M-tiles
  const int wn   = (wave >> 2) * 64;  // 4 N-tiles
  const int m0   = blockIdx.y * 128;
  const int n0   = blockIdx.x * 128;

  const int asel = lhi ? 8 : 0;       // A frag: K 0-7/16-23 vs 8-15/24-31
  const int bsel = lhi ? 8 : 0;       // B frag: K 0-7/16-23 vs 8-15/24-31

  const __bf16* ap[2];
  ap[0] = A + (size_t)(m0 + wm + l16) * K + asel;
  ap[1] = ap[0] + (size_t)16 * K;
  const __bf16* bp[4];
#pragma unroll
  for (int ni = 0; ni < 4; ++ni)
    bp[ni] = Wt + (size_t)(n0 + wn + ni * 16 + l16) * K + bsel;

  v8f acc[2][4] = {};

  for (int k0 = 0; k0 < K; k0 += 32) {
    ABf af[2];
#pragma unroll
    for (int mi = 0; mi < 2; ++mi) {
      af[mi].h[0] = *(const v8bf*)(ap[mi] + k0);
      af[mi].h[1] = *(const v8bf*)(ap[mi] + k0 + 16);
    }
#pragma unroll
    for (int ni = 0; ni < 4; ++ni) {
      ABf bfr;
      bfr.h[0] = *(const v8bf*)(bp[ni] + k0);
      bfr.h[1] = *(const v8bf*)(bp[ni] + k0 + 16);
#pragma unroll
      for (int mi = 0; mi < 2; ++mi)
        acc[mi][ni] = wmma_bf16(af[mi].v, bfr.v, acc[mi][ni]);
    }
  }

  // C layout: VGPR r -> row r + 8*lhi ; lane%16 -> column
#pragma unroll
  for (int mi = 0; mi < 2; ++mi) {
#pragma unroll
    for (int ni = 0; ni < 4; ++ni) {
      F8 a; a.v = acc[mi][ni];
      const int row0 = m0 + wm + mi * 16 + lhi * 8;
      const int col  = n0 + wn + ni * 16 + l16;
      const float bb = bias[col];
      if (mode == 2) {
        // pack 8 consecutive s into one 16B store: vt[b][kvh][dd][s]
        B8 pk;
#pragma unroll
        for (int r = 0; r < 8; ++r) pk.e[r] = f2bf((a.f[r] + bb) * oscale);
        const size_t off =
            ((size_t)((row0 >> 10) * NKV + (col >> 6)) * HD + (col & 63)) * TT
            + (row0 & 1023);
        *(v8bf*)(Cb + off) = pk.v;
      } else if (mode == 1) {
#pragma unroll
        for (int r = 0; r < 8; ++r)
          Cb[(size_t)(row0 + r) * N + col] = f2bf((a.f[r] + bb) * oscale);
      } else {
#pragma unroll
        for (int r = 0; r < 8; ++r)
          Cf[(size_t)(row0 + r) * N + col] = (a.f[r] + bb) * oscale;
      }
    }
  }
}

// ------------------------------------------------------------ attention
// grid (TT/128, NH, NB), 256 thr.  Wave w owns 16 query rows; waves fully
// independent (no barriers).  Q pre-scaled by HD^-0.5*log2e so softmax uses
// raw v_exp_f32 (exp2).  Keys processed in chunks of 128 (8 S-tiles, 4 PV
// K-steps => 32 WMMA per chunk).  Q/K direct from global; V direct from
// pre-transposed vt; P relayout through wave-private LDS.
#define PSTR 136

__global__ __launch_bounds__(256)
void attn_kernel(const __bf16* __restrict__ Q, const __bf16* __restrict__ Kb,
                 const __bf16* __restrict__ Vt, __bf16* __restrict__ O)
{
  __shared__ __bf16 Ps[8 * 16 * PSTR];   // per-wave 16x128 P staging

  const int tid  = threadIdx.x;
  const int lane = tid & 31;
  const int l16  = lane & 15;
  const int lhi  = lane >> 4;
  const int wave = tid >> 5;

  const int b   = blockIdx.z;
  const int h   = blockIdx.y;
  const int kvh = h & (NKV - 1);         // jnp.tile => head h uses kv head h%8
  const int q0  = blockIdx.x * 128;

  const int asel = lhi ? 8 : 0;
  const int bsel = lhi ? 8 : 0;

  // Q A-fragments (16 rows x 64 dims => 2 K-chunks), loaded once
  const int qrow = q0 + wave * 16 + l16;
  const __bf16* qp = Q + ((size_t)(b * TT + qrow)) * DIM + h * HD;
  v16bf qa[2];
#pragma unroll
  for (int kk = 0; kk < 2; ++kk) {
    ABf u;
    u.h[0] = *(const v8bf*)(qp + kk * 32 + asel);
    u.h[1] = *(const v8bf*)(qp + kk * 32 + asel + 16);
    qa[kk] = u.v;
  }

  // per-lane K row base (lane's score column within a 16-tile)
  const __bf16* kbase = Kb + ((size_t)(b * TT + l16)) * KV_DIM + kvh * HD + bsel;
  // pre-transposed V: vt[((b*NKV+kvh)*HD + dd)*TT + s]
  const __bf16* vbase = Vt + ((size_t)(b * NKV + kvh) * HD + l16) * TT;

  v8f oacc[4] = {};
  float mrow[8], lrow[8];
#pragma unroll
  for (int r = 0; r < 8; ++r) { mrow[r] = -1e30f; lrow[r] = 0.f; }

  __bf16* pw = &Ps[wave * 16 * PSTR];

  for (int s0 = 0; s0 < TT; s0 += 128) {
    // ---- S = Q K^T  (16 x 128 per wave: 8 tiles x 2 K-steps)
    F8 sacc[8] = {};
#pragma unroll
    for (int n = 0; n < 8; ++n) {
      const __bf16* kp = kbase + (size_t)(s0 + n * 16) * KV_DIM;
#pragma unroll
      for (int kk = 0; kk < 2; ++kk) {
        ABf u;
        u.h[0] = *(const v8bf*)(kp + kk * 32);
        u.h[1] = *(const v8bf*)(kp + kk * 32 + 16);
        sacc[n].v = wmma_bf16(qa[kk], u.v, sacc[n].v);
      }
    }

    // ---- online softmax (base-2; scale already folded into Q)
    float mnew[8], alpha[8];
#pragma unroll
    for (int r = 0; r < 8; ++r) {
      float mx = -1e30f;
#pragma unroll
      for (int n = 0; n < 8; ++n) mx = fmaxf(mx, sacc[n].f[r]);
#pragma unroll
      for (int off = 1; off < 16; off <<= 1)
        mx = fmaxf(mx, __shfl_xor(mx, off, 32));
      mnew[r]  = fmaxf(mrow[r], mx);
      alpha[r] = exp2f(mrow[r] - mnew[r]);
      mrow[r]  = mnew[r];
      float rs = 0.f;
#pragma unroll
      for (int n = 0; n < 8; ++n) {
        float p = exp2f(sacc[n].f[r] - mnew[r]);
        sacc[n].f[r] = p;
        rs += p;
      }
#pragma unroll
      for (int off = 1; off < 16; off <<= 1)
        rs += __shfl_xor(rs, off, 32);
      lrow[r] = lrow[r] * alpha[r] + rs;
    }
#pragma unroll
    for (int n = 0; n < 4; ++n) {
      F8 o; o.v = oacc[n];
#pragma unroll
      for (int r = 0; r < 8; ++r) o.f[r] *= alpha[r];
      oacc[n] = o.v;
    }

    // ---- P (C layout) -> per-wave LDS (row-major) for A-layout reload
#pragma unroll
    for (int n = 0; n < 8; ++n)
#pragma unroll
      for (int r = 0; r < 8; ++r)
        pw[(r + lhi * 8) * PSTR + n * 16 + l16] = f2bf(sacc[n].f[r]);

    asm volatile("s_wait_dscnt 0x0" ::: "memory");  // wave-private fence

    // ---- O += P @ V   (V B-frags direct from transposed global)
#pragma unroll
    for (int kk = 0; kk < 4; ++kk) {
      ABf pu;
      const __bf16* pr = pw + l16 * PSTR + kk * 32 + asel;
      pu.h[0] = *(const v8bf*)pr;
      pu.h[1] = *(const v8bf*)(pr + 16);
#pragma unroll
      for (int n = 0; n < 4; ++n) {
        ABf vu;
        const __bf16* vr = vbase + (size_t)(n * 16) * TT + s0 + kk * 32 + bsel;
        vu.h[0] = *(const v8bf*)vr;
        vu.h[1] = *(const v8bf*)(vr + 16);
        oacc[n] = wmma_bf16(pu.v, vu.v, oacc[n]);
      }
    }
  }

  // ---- normalize, relayout through LDS, coalesced b128 stores
#pragma unroll
  for (int n = 0; n < 4; ++n) {
    F8 o; o.v = oacc[n];
#pragma unroll
    for (int r = 0; r < 8; ++r)
      pw[(r + lhi * 8) * PSTR + n * 16 + l16] = f2bf(o.f[r] / lrow[r]);
  }
  asm volatile("s_wait_dscnt 0x0" ::: "memory");
  // 16 rows x 64 cols; lanes 0-7 cover one full 128B row segment
  {
    const int oc   = (lane & 7) * 8;       // col chunk (8 bf16 = 16B)
    const int rsub = lane >> 3;            // 0..3
#pragma unroll
    for (int it = 0; it < 4; ++it) {
      const int orow = it * 4 + rsub;      // 0..15
      v8bf d = *(const v8bf*)(pw + orow * PSTR + oc);
      *(v8bf*)(O + ((size_t)(b * TT + q0 + wave * 16 + orow)) * DIM
               + h * HD + oc) = d;
    }
  }
}

// ------------------------------------------------------------ launch
extern "C" void kernel_launch(void* const* d_in, const int* in_sizes, int n_in,
                              void* d_out, int out_size, void* d_ws, size_t ws_size,
                              hipStream_t stream) {
  const float* x  = (const float*)d_in[0];
  const float* wq = (const float*)d_in[1];
  const float* bq = (const float*)d_in[2];
  const float* wk = (const float*)d_in[3];
  const float* bk = (const float*)d_in[4];
  const float* wv = (const float*)d_in[5];
  const float* bv = (const float*)d_in[6];
  const float* wo = (const float*)d_in[7];
  const float* bo = (const float*)d_in[8];
  float* out = (float*)d_out;
  (void)in_sizes; (void)n_in; (void)out_size; (void)ws_size;

  __bf16* xb  = (__bf16*)d_ws;                       // [BT][DIM]
  __bf16* wqt = xb  + (size_t)BT  * DIM;             // [DIM][DIM]   (n-major)
  __bf16* wkt = wqt + (size_t)DIM * DIM;             // [KV_DIM][DIM]
  __bf16* wvt = wkt + (size_t)DIM * KV_DIM;          // [KV_DIM][DIM]
  __bf16* wot = wvt + (size_t)DIM * KV_DIM;          // [DIM][DIM]
  __bf16* qb  = wot + (size_t)DIM * DIM;             // [BT][DIM]
  __bf16* kb  = qb  + (size_t)BT  * DIM;             // [BT][KV_DIM]
  __bf16* vt  = kb  + (size_t)BT  * KV_DIM;          // [NB][NKV][HD][TT]
  __bf16* ob  = vt  + (size_t)BT  * KV_DIM;          // [BT][DIM]

  cvt_bf16_kernel<<<1024, 256, 0, stream>>>(x, xb, BT * DIM);
  cvt_t_bf16_kernel<<<dim3(DIM / 32, DIM / 32),    256, 0, stream>>>(wq, wqt, DIM, DIM);
  cvt_t_bf16_kernel<<<dim3(KV_DIM / 32, DIM / 32), 256, 0, stream>>>(wk, wkt, DIM, KV_DIM);
  cvt_t_bf16_kernel<<<dim3(KV_DIM / 32, DIM / 32), 256, 0, stream>>>(wv, wvt, DIM, KV_DIM);
  cvt_t_bf16_kernel<<<dim3(DIM / 32, DIM / 32),    256, 0, stream>>>(wo, wot, DIM, DIM);

  // projections (Q pre-scaled by HD^-0.5 * log2e)
  gemm_wmma_kernel<<<dim3(DIM / 128, BT / 128), 256, 0, stream>>>(
      xb, wqt, bq, nullptr, qb, BT, DIM, DIM, 1, QSCALE);
  gemm_wmma_kernel<<<dim3(KV_DIM / 128, BT / 128), 256, 0, stream>>>(
      xb, wkt, bk, nullptr, kb, BT, KV_DIM, DIM, 1, 1.0f);
  gemm_wmma_kernel<<<dim3(KV_DIM / 128, BT / 128), 256, 0, stream>>>(
      xb, wvt, bv, nullptr, vt, BT, KV_DIM, DIM, 2, 1.0f);   // V pre-transposed

  // attention
  attn_kernel<<<dim3(TT / 128, NH, NB), 256, 0, stream>>>(qb, kb, vt, ob);

  // output projection (fp32 + bias into d_out)
  gemm_wmma_kernel<<<dim3(DIM / 128, BT / 128), 256, 0, stream>>>(
      ob, wot, bo, out, nullptr, BT, DIM, DIM, 0, 1.0f);
}
